// LlamaStyleMoEFFN_7602092114211
// MI455X (gfx1250) — matrix operand order repaired and checked by reading, not compiled, not run on hardware
//
#include <hip/hip_runtime.h>
#include <hip/hip_bf16.h>

#define DMODEL 1024
#define DFF    2048
#define NEXP   8
#define NTOK   4096
#define MT     32      // token-tile rows per block
#define NT     128     // output cols per block
#define KT     32      // K step (bf16 WMMA depth)

typedef __attribute__((ext_vector_type(16))) __bf16 bf16x16;
typedef __attribute__((ext_vector_type(8)))  float  floatx8;
typedef __attribute__((ext_vector_type(4)))  unsigned int uint32x4;
typedef __attribute__((ext_vector_type(8)))  int int32x8;
typedef __attribute__((ext_vector_type(4)))  int int32x4;

union FragBF { bf16x16 v; uint4 q[2]; };

#if defined(__has_builtin)
#if __has_builtin(__builtin_amdgcn_tensor_load_to_lds)
#define HAVE_TDM 1
#endif
#if __has_builtin(__builtin_amdgcn_s_wait_tensorcnt)
#define HAVE_TWAIT 1
#endif
#endif

__device__ __forceinline__ unsigned short f2bf(float f) {
  union { float f; unsigned int u; } v; v.f = f;
  unsigned int r = v.u + 0x7FFFu + ((v.u >> 16) & 1u);   // round-to-nearest-even
  return (unsigned short)(r >> 16);
}

// CDNA5 async global->LDS copy (8 bytes per lane), ASYNCcnt-tracked.
__device__ __forceinline__ void async_copy_b64(void* lds_dst, const void* gsrc) {
  unsigned lds_off = (unsigned)(unsigned long long)lds_dst;  // low 32 bits = LDS offset
  asm volatile("global_load_async_to_lds_b64 %0, %1, off"
               :: "v"(lds_off), "v"(gsrc) : "memory");
}
__device__ __forceinline__ void wait_async0() {
  asm volatile("s_wait_asynccnt 0x0" ::: "memory");
}
__device__ __forceinline__ void wait_tensor0() {
#ifdef HAVE_TWAIT
  __builtin_amdgcn_s_wait_tensorcnt(0);
#else
  asm volatile("s_wait_tensorcnt 0x0" ::: "memory");
#endif
}

#ifdef HAVE_TDM
// TDM 2D tile load: rows x tile_k bf16 elements, packed into LDS.
// D# packing per CDNA5 ISA.
__device__ __forceinline__ void tdm_load_2d(unsigned lds_addr, const void* gaddr,
                                            unsigned tensor_k, unsigned tensor_rows,
                                            unsigned tile_k, unsigned tile_rows,
                                            unsigned row_stride_elems) {
  unsigned long long ga = (unsigned long long)gaddr;
  uint32x4 g0;
  g0.x = 1u;                                              // count=1, user mode
  g0.y = lds_addr;                                        // lds_addr [63:32]
  g0.z = (unsigned)(ga & 0xFFFFFFFFu);                    // global_addr lo
  g0.w = (unsigned)((ga >> 32) & 0x01FFFFFFu) | (2u << 30); // global_addr hi, type=2
  int32x8 g1;
  g1[0] = (int)(1u << 16);                                // data_size=1 (2 bytes)
  g1[1] = (int)((tensor_k & 0xFFFFu) << 16);              // tensor_dim0[15:0]
  g1[2] = (int)(((tensor_k >> 16) & 0xFFFFu) | ((tensor_rows & 0xFFFFu) << 16));
  g1[3] = (int)(((tensor_rows >> 16) & 0xFFFFu) | ((tile_k & 0xFFFFu) << 16)); // tile_dim0
  g1[4] = (int)(tile_rows & 0xFFFFu);                     // tile_dim1 (tile_dim2=0)
  g1[5] = (int)row_stride_elems;                          // tensor_dim0_stride lo
  g1[6] = 0;
  g1[7] = 0;
  int32x4 z4 = {0, 0, 0, 0};
  int32x8 z8 = {0, 0, 0, 0, 0, 0, 0, 0};
  __builtin_amdgcn_tensor_load_to_lds(g0, g1, z4, z4, z8, 0);
}
#endif

__device__ __forceinline__ unsigned lds_off32(const void* p) {
  return (unsigned)(unsigned long long)p;
}

__device__ __forceinline__ float fast_silu(float g) {
  return g * __builtin_amdgcn_rcpf(1.0f + __expf(-g));
}

// ---------------- utility kernels ----------------
__global__ void zero_f32_k(float* __restrict__ p, int n) {
  int i = (blockIdx.x * blockDim.x + threadIdx.x) * 4;
  if (i < n) *(float4*)(p + i) = make_float4(0.f, 0.f, 0.f, 0.f);
}

__global__ void zero_i32_k(int* __restrict__ p, int n) {
  int i = blockIdx.x * blockDim.x + threadIdx.x;
  if (i < n) p[i] = 0;
}

__global__ void conv_x_k(const float* __restrict__ x, unsigned short* __restrict__ xb, int n) {
  int i = (blockIdx.x * blockDim.x + threadIdx.x) * 4;
  if (i < n) {
    float4 v = *(const float4*)(x + i);
    uint2 o;
    o.x = (unsigned)f2bf(v.x) | ((unsigned)f2bf(v.y) << 16);
    o.y = (unsigned)f2bf(v.z) | ((unsigned)f2bf(v.w) << 16);
    *(uint2*)(xb + i) = o;
  }
}

// transpose + fp32->bf16: wt[e][n][k] = bf16(w[e][k][n]); one-time per launch
__global__ __launch_bounds__(256) void convT_k(const float* __restrict__ w,
                                               unsigned short* __restrict__ wt,
                                               int K, int N) {
  __shared__ float t[32][33];
  int e  = blockIdx.z;
  int k0 = blockIdx.y * 32;
  int n0 = blockIdx.x * 32;
  const float* we = w + (size_t)e * K * N;
  unsigned short* wte = wt + (size_t)e * K * N;
  int tx = threadIdx.x & 31;
  int ty = threadIdx.x >> 5;
#pragma unroll
  for (int p = 0; p < 4; ++p) {
    int k = ty + p * 8;
    t[k][tx] = we[(size_t)(k0 + k) * N + n0 + tx];
  }
  __syncthreads();
#pragma unroll
  for (int p = 0; p < 4; ++p) {
    int n = ty + p * 8;
    wte[(size_t)(n0 + n) * K + k0 + tx] = f2bf(t[tx][n]);
  }
}

// ---------------- router: softmax + top-2 + append to expert lists ----------------
__global__ __launch_bounds__(256) void moe_router_k(
    const float* __restrict__ x, const float* __restrict__ rw,
    int* __restrict__ counts, int* __restrict__ idx_list, float* __restrict__ wgt_list) {
  int wave = threadIdx.x >> 5;
  int lane = threadIdx.x & 31;
  int n = blockIdx.x * 8 + wave;
  if (n >= NTOK) return;

  float acc[NEXP];
#pragma unroll
  for (int e = 0; e < NEXP; ++e) acc[e] = 0.f;

  const float* xr = x + (size_t)n * DMODEL;
  for (int d = lane; d < DMODEL; d += 32) {
    float xv = xr[d];
#pragma unroll
    for (int e = 0; e < NEXP; ++e) acc[e] += xv * rw[e * DMODEL + d];
  }
#pragma unroll
  for (int e = 0; e < NEXP; ++e) {
    float v = acc[e];
    for (int off = 16; off > 0; off >>= 1) v += __shfl_xor(v, off, 32);
    acc[e] = v;
  }
  float mx = acc[0];
#pragma unroll
  for (int e = 1; e < NEXP; ++e) mx = fmaxf(mx, acc[e]);
  float p[NEXP], se = 0.f;
#pragma unroll
  for (int e = 0; e < NEXP; ++e) { p[e] = __expf(acc[e] - mx); se += p[e]; }
  float inv_se = __builtin_amdgcn_rcpf(se);
#pragma unroll
  for (int e = 0; e < NEXP; ++e) p[e] *= inv_se;

  int i1 = 0;
#pragma unroll
  for (int e = 1; e < NEXP; ++e) if (p[e] > p[i1]) i1 = e;
  int i2 = (i1 == 0) ? 1 : 0;
#pragma unroll
  for (int e = 0; e < NEXP; ++e) if (e != i1 && p[e] > p[i2]) i2 = e;

  float invd = __builtin_amdgcn_rcpf(p[i1] + p[i2] + 1e-9f);
  if (lane == 0) {
    int s1 = atomicAdd(&counts[i1], 1);
    idx_list[i1 * NTOK + s1] = n; wgt_list[i1 * NTOK + s1] = p[i1] * invd;
    int s2 = atomicAdd(&counts[i2], 1);
    idx_list[i2 * NTOK + s2] = n; wgt_list[i2 * NTOK + s2] = p[i2] * invd;
  }
}

__global__ void prefix_k(const int* __restrict__ counts, int* __restrict__ slot_base) {
  if (threadIdx.x == 0 && blockIdx.x == 0) {
    int b = 0;
#pragma unroll
    for (int e = 0; e < NEXP; ++e) { slot_base[e] = b; b += counts[e]; }
    slot_base[NEXP] = b;
  }
}

// ---------------- GEMM1: h = silu(x@gate_w + gate_b) * (x@up_w + up_b) ----------------
__global__ __launch_bounds__(256) void moe_gemm1_k(
    const unsigned short* __restrict__ xb,
    const unsigned short* __restrict__ gwT, const unsigned short* __restrict__ uwT,
    const float* __restrict__ gate_b, const float* __restrict__ up_b,
    const int* __restrict__ idx_list, const int* __restrict__ counts,
    const int* __restrict__ slot_base, unsigned short* __restrict__ hbuf) {
  int e  = blockIdx.z;
  int m0 = blockIdx.y * MT;
  int f0 = blockIdx.x * NT;
  int cnt = counts[e];
  if (m0 >= cnt) return;

  __shared__ unsigned short As[2][MT][KT];   //  4 KB
  __shared__ unsigned short Bg[2][NT][KT];   // 16 KB (row n, contiguous k)
  __shared__ unsigned short Bu[2][NT][KT];   // 16 KB
  __shared__ int tIdx[MT];

  int tid = threadIdx.x;
  if (tid < MT) {
    int r = m0 + tid;
    int rr = (r < cnt) ? r : (cnt - 1);
    tIdx[tid] = idx_list[e * NTOK + rr];
  }
  __syncthreads();

  int lane = tid & 31, wave = tid >> 5;
  int wm = wave >> 2, wn = wave & 3;
  int hi = lane >> 4, l16 = lane & 15;

  floatx8 accg[2] = {{}, {}};
  floatx8 accu[2] = {{}, {}};

  const unsigned short* gwe = gwT + (size_t)e * DFF * DMODEL;
  const unsigned short* uwe = uwT + (size_t)e * DFF * DMODEL;

  int am = tid >> 3;
  int ak = (tid & 7) * 4;
  const unsigned short* arow = xb + (size_t)tIdx[am] * DMODEL + ak;

  // B staging: all-bf16, pure data movement into LDS
  auto stage_b = [&](int buf, int k0) {
#ifdef HAVE_TDM
    if (wave == 0) {   // one DMA descriptor per matrix tile
      tdm_load_2d(lds_off32(&Bg[buf][0][0]), gwe + (size_t)f0 * DMODEL + k0,
                  DMODEL, DFF, KT, NT, DMODEL);
      tdm_load_2d(lds_off32(&Bu[buf][0][0]), uwe + (size_t)f0 * DMODEL + k0,
                  DMODEL, DFF, KT, NT, DMODEL);
    }
#else
#pragma unroll
    for (int j = 0; j < 4; ++j) {
      int c = tid + j * 256;
      int row = c >> 3, off = (c & 7) * 4;
      async_copy_b64(&Bg[buf][row][off], gwe + (size_t)(f0 + row) * DMODEL + k0 + off);
      async_copy_b64(&Bu[buf][row][off], uwe + (size_t)(f0 + row) * DMODEL + k0 + off);
    }
#endif
  };
  auto wait_stage = [&]() {
#ifdef HAVE_TDM
    if (wave == 0) wait_tensor0();
#endif
    wait_async0();
  };

  const int NSTEP = DMODEL / KT;

  // prologue: stage step 0 into buffer 0
  async_copy_b64(&As[0][am][ak], arow);
  stage_b(0, 0);
  wait_stage();
  __syncthreads();

#pragma unroll 2
  for (int it = 0; it < NSTEP; ++it) {
    int cur = it & 1, nxt = cur ^ 1;
    bool hasNext = (it + 1 < NSTEP);
    int k0n = (it + 1) * KT;

    if (hasNext) {
      async_copy_b64(&As[nxt][am][ak], arow + k0n);   // gathered A rows
      stage_b(nxt, k0n);
    }

    FragBF a;
    int m = wm * 16 + l16;
    a.q[0] = *(const uint4*)&As[cur][m][hi * 16];
    a.q[1] = *(const uint4*)&As[cur][m][hi * 16 + 8];
#pragma unroll
    for (int s = 0; s < 2; ++s) {
      int n = wn * 32 + s * 16 + l16;
      FragBF bg, bu;
      bg.q[0] = *(const uint4*)&Bg[cur][n][hi * 16];
      bg.q[1] = *(const uint4*)&Bg[cur][n][hi * 16 + 8];
      bu.q[0] = *(const uint4*)&Bu[cur][n][hi * 16];
      bu.q[1] = *(const uint4*)&Bu[cur][n][hi * 16 + 8];
      accg[s] = __builtin_amdgcn_wmma_f32_16x16x32_bf16(false, a.v, false, bg.v, (short)0, accg[s], false, false);
      accu[s] = __builtin_amdgcn_wmma_f32_16x16x32_bf16(false, a.v, false, bu.v, (short)0, accu[s], false, false);
    }

    if (hasNext) wait_stage();
    __syncthreads();
  }

  int sb = slot_base[e];
#pragma unroll
  for (int s = 0; s < 2; ++s) {
    int n_g = f0 + wn * 32 + s * 16 + l16;
    float gb = gate_b[e * DFF + n_g];
    float ub = up_b[e * DFF + n_g];
    float ga[8], ua[8];
#pragma unroll
    for (int r = 0; r < 8; ++r) { ga[r] = accg[s][r]; ua[r] = accu[s][r]; }
#pragma unroll
    for (int r = 0; r < 8; ++r) {
      int ml  = wm * 16 + r + hi * 8;
      int row = m0 + ml;
      if (row < cnt) {
        float h = fast_silu(ga[r] + gb) * (ua[r] + ub);
        hbuf[(size_t)(sb + row) * DFF + n_g] = f2bf(h);
      }
    }
  }
}

// ---------------- GEMM2: out[token] += w * (h @ down_w + down_b) ----------------
__global__ __launch_bounds__(256) void moe_gemm2_k(
    const unsigned short* __restrict__ hbuf,
    const unsigned short* __restrict__ dwT, const float* __restrict__ down_b,
    const int* __restrict__ idx_list, const float* __restrict__ wgt_list,
    const int* __restrict__ counts, const int* __restrict__ slot_base,
    float* __restrict__ out) {
  int e  = blockIdx.z;
  int m0 = blockIdx.y * MT;
  int d0 = blockIdx.x * NT;
  int cnt = counts[e];
  if (m0 >= cnt) return;

  __shared__ unsigned short As[2][MT][KT];   //  4 KB
  __shared__ unsigned short Bd[2][NT][KT];   // 16 KB
  __shared__ int   tIdx[MT];
  __shared__ float tW[MT];

  int tid = threadIdx.x;
  if (tid < MT) {
    int r = m0 + tid;
    int rr = (r < cnt) ? r : (cnt - 1);
    tIdx[tid] = idx_list[e * NTOK + rr];
    tW[tid]   = wgt_list[e * NTOK + rr];
  }
  __syncthreads();

  int lane = tid & 31, wave = tid >> 5;
  int wm = wave >> 2, wn = wave & 3;
  int hi = lane >> 4, l16 = lane & 15;

  floatx8 acc[2] = {{}, {}};
  const unsigned short* dwe = dwT + (size_t)e * DMODEL * DFF;
  int sb = slot_base[e];

  int am = tid >> 3;
  int ak = (tid & 7) * 4;
  const unsigned short* arow = hbuf + (size_t)(sb + m0 + am) * DFF + ak;

  auto stage_b = [&](int buf, int k0) {
#ifdef HAVE_TDM
    if (wave == 0) {
      tdm_load_2d(lds_off32(&Bd[buf][0][0]), dwe + (size_t)d0 * DFF + k0,
                  DFF, DMODEL, KT, NT, DFF);
    }
#else
#pragma unroll
    for (int j = 0; j < 4; ++j) {
      int c = tid + j * 256;
      int row = c >> 3, off = (c & 7) * 4;
      async_copy_b64(&Bd[buf][row][off], dwe + (size_t)(d0 + row) * DFF + k0 + off);
    }
#endif
  };
  auto wait_stage = [&]() {
#ifdef HAVE_TDM
    if (wave == 0) wait_tensor0();
#endif
    wait_async0();
  };

  const int NSTEP = DFF / KT;

  async_copy_b64(&As[0][am][ak], arow);
  stage_b(0, 0);
  wait_stage();
  __syncthreads();

#pragma unroll 2
  for (int it = 0; it < NSTEP; ++it) {
    int cur = it & 1, nxt = cur ^ 1;
    bool hasNext = (it + 1 < NSTEP);
    int k0n = (it + 1) * KT;

    if (hasNext) {
      async_copy_b64(&As[nxt][am][ak], arow + k0n);
      stage_b(nxt, k0n);
    }

    FragBF a;
    int m = wm * 16 + l16;
    a.q[0] = *(const uint4*)&As[cur][m][hi * 16];
    a.q[1] = *(const uint4*)&As[cur][m][hi * 16 + 8];
#pragma unroll
    for (int s = 0; s < 2; ++s) {
      int n = wn * 32 + s * 16 + l16;
      FragBF b;
      b.q[0] = *(const uint4*)&Bd[cur][n][hi * 16];
      b.q[1] = *(const uint4*)&Bd[cur][n][hi * 16 + 8];
      acc[s] = __builtin_amdgcn_wmma_f32_16x16x32_bf16(false, a.v, false, b.v, (short)0, acc[s], false, false);
    }

    if (hasNext) wait_stage();
    __syncthreads();
  }

#pragma unroll
  for (int s = 0; s < 2; ++s) {
    int n_g = d0 + wn * 32 + s * 16 + l16;
    float db = down_b[e * DMODEL + n_g];
    float ra[8];
#pragma unroll
    for (int r = 0; r < 8; ++r) ra[r] = acc[s][r];
#pragma unroll
    for (int r = 0; r < 8; ++r) {
      int ml  = wm * 16 + r + hi * 8;
      int row = m0 + ml;
      if (row < cnt) {
        int   t = tIdx[ml];
        float w = tW[ml];
        atomicAdd(&out[(size_t)t * DMODEL + n_g], w * (ra[r] + db));
      }
    }
  }
}

// ---------------- launch ----------------
extern "C" void kernel_launch(void* const* d_in, const int* in_sizes, int n_in,
                              void* d_out, int out_size, void* d_ws, size_t ws_size,
                              hipStream_t stream) {
  (void)in_sizes; (void)n_in; (void)out_size; (void)ws_size;
  const float* x        = (const float*)d_in[0];
  const float* router_w = (const float*)d_in[1];
  const float* gate_w   = (const float*)d_in[2];
  const float* gate_b   = (const float*)d_in[3];
  const float* up_w     = (const float*)d_in[4];
  const float* up_b     = (const float*)d_in[5];
  const float* down_w   = (const float*)d_in[6];
  const float* down_b   = (const float*)d_in[7];
  float* out = (float*)d_out;

  char* ws = (char*)d_ws;
  int*   counts    = (int*)ws;                        // 8 ints
  int*   slot_base = (int*)(ws + 64);                 // 9 ints
  int*   idx_list  = (int*)(ws + 256);                // 128 KB
  float* wgt_list  = (float*)(ws + 256 + 131072);     // 128 KB
  const size_t WEXP = (size_t)DMODEL * DFF;           // 2 M elements per expert matrix
  unsigned short* xb   = (unsigned short*)(ws + 524288);              // 8 MB
  unsigned short* hbuf = xb + (size_t)NTOK * DMODEL;                  // ~33.7 MB
  unsigned short* gwT  = hbuf + (size_t)(2 * NTOK + MT) * DFF;        // 32 MB
  unsigned short* uwT  = gwT + NEXP * WEXP;                           // 32 MB
  unsigned short* dwT  = uwT + NEXP * WEXP;                           // 32 MB

  const int NXE = NTOK * DMODEL;  // 4194304

  zero_i32_k<<<1, 32, 0, stream>>>(counts, NEXP);
  zero_f32_k<<<NXE / 1024, 256, 0, stream>>>(out, NXE);
  conv_x_k<<<NXE / 1024, 256, 0, stream>>>(x, xb, NXE);
  // one-time: bf16 + transpose of all expert weights
  convT_k<<<dim3(DFF / 32, DMODEL / 32, NEXP), 256, 0, stream>>>(gate_w, gwT, DMODEL, DFF);
  convT_k<<<dim3(DFF / 32, DMODEL / 32, NEXP), 256, 0, stream>>>(up_w,   uwT, DMODEL, DFF);
  convT_k<<<dim3(DMODEL / 32, DFF / 32, NEXP), 256, 0, stream>>>(down_w, dwT, DFF, DMODEL);
  moe_router_k<<<NTOK / 8, 256, 0, stream>>>(x, router_w, counts, idx_list, wgt_list);
  prefix_k<<<1, 32, 0, stream>>>(counts, slot_base);
  moe_gemm1_k<<<dim3(DFF / NT, NTOK / MT, NEXP), 256, 0, stream>>>(
      xb, gwT, uwT, gate_b, up_b, idx_list, counts, slot_base, hbuf);
  moe_gemm2_k<<<dim3(DMODEL / NT, NTOK / MT, NEXP), 256, 0, stream>>>(
      hbuf, dwT, down_b, idx_list, wgt_list, counts, slot_base, out);
}
